// DenseExpert_35785667510883
// MI455X (gfx1250) — matrix operand order repaired and checked by reading, not compiled, not run on hardware
//
#include <hip/hip_runtime.h>

// ---------------------------------------------------------------------------
// DenseExpert: r[b,u] = sum_e g[b,e] * (x[b,:] . alpha[e,u,:]) + sum_e g[b,e]*beta[e,u]
// One bf16 GEMM, fp32 accumulate:
//   A[b, e*512+d] = g[b,e] * x[b,d]   (gated + converted during LDS staging)
//   Bm[e*512+d,u] = alpha[e,u,d]      (alpha rows contiguous in d)
//   bias folded in as one zero-padded K-tile: A'=g, B'=beta
// MI455X: 34.4 GFLOP vs ~42MB unique traffic -> compute-bound; bf16 WMMA path.
// x (16.8MB) is re-read 8x across experts but lives in the 192MB L2.
// ---------------------------------------------------------------------------

typedef __attribute__((ext_vector_type(16))) __bf16 bf16x16;
typedef __attribute__((ext_vector_type(8)))  __bf16 bf16x8;
typedef __attribute__((ext_vector_type(4)))  __bf16 bf16x4;
typedef __attribute__((ext_vector_type(8)))  float  floatx8;
typedef __attribute__((ext_vector_type(4)))  float  floatx4;
typedef unsigned short u16;

#define BB  8192   // batch (GEMM M)
#define EE  8      // experts
#define UU  512    // units  (GEMM N)
#define DD  512    // depth  (per-expert K)

#define BM  128
#define BN  64
#define BK  64
#define LDA 72     // LDS row stride (u16) = BK + 8 pad; keeps 16B chunk alignment
#define LDB 72
#define NSTEPS 65  // step 0 = bias tile (K padded 8->64), steps 1..64 = main K loop

// Stage one 128x64 A tile (gated x -> bf16) and one 64x64 B tile (alpha -> bf16).
// Each thread handles 8 contiguous elements per iteration: 2x b128 global loads,
// packed f32->bf16 converts, one 16B LDS store.
__device__ __forceinline__ void stage_tiles(
    u16* __restrict__ As, u16* __restrict__ Bs,
    const float* __restrict__ x, const float* __restrict__ g,
    const float* __restrict__ alpha, const float* __restrict__ beta,
    int blockM, int blockN, int step, int tid)
{
  if (step == 0) {
    // Bias tile: A[m][k] = (k<8) ? g[m][k] : 0 ; B[n][k] = (k<8) ? beta[k][n] : 0
#pragma unroll
    for (int i = 0; i < 4; ++i) {
      int idx = tid + i * 256;          // 1024 chunks = 128 rows x 8
      int row = idx >> 3;
      int c   = (idx & 7) << 3;
      floatx4 v0 = {}, v1 = {};
      if (c == 0) {
        const float* gp = &g[(size_t)(blockM + row) * EE];
        v0 = *reinterpret_cast<const floatx4*>(gp);
        v1 = *reinterpret_cast<const floatx4*>(gp + 4);
      }
      bf16x4 h0 = __builtin_convertvector(v0, bf16x4);
      bf16x4 h1 = __builtin_convertvector(v1, bf16x4);
      *reinterpret_cast<bf16x8*>(&As[row * LDA + c]) =
          __builtin_shufflevector(h0, h1, 0, 1, 2, 3, 4, 5, 6, 7);
    }
#pragma unroll
    for (int i = 0; i < 2; ++i) {
      int idx = tid + i * 256;          // 512 chunks = 64 rows x 8
      int n   = idx >> 3;
      int c   = (idx & 7) << 3;
      floatx4 v0 = {}, v1 = {};
      if (c == 0) {
        int gn = blockN + n;
        v0 = floatx4{beta[0 * UU + gn], beta[1 * UU + gn],
                     beta[2 * UU + gn], beta[3 * UU + gn]};
        v1 = floatx4{beta[4 * UU + gn], beta[5 * UU + gn],
                     beta[6 * UU + gn], beta[7 * UU + gn]};
      }
      bf16x4 h0 = __builtin_convertvector(v0, bf16x4);
      bf16x4 h1 = __builtin_convertvector(v1, bf16x4);
      *reinterpret_cast<bf16x8*>(&Bs[n * LDB + c]) =
          __builtin_shufflevector(h0, h1, 0, 1, 2, 3, 4, 5, 6, 7);
    }
  } else {
    const int k0 = (step - 1) * BK;
    const int e  = k0 >> 9;             // expert for this K-tile (BK | 512)
    const int d0 = k0 & 511;
    const int k1 = k0 + BK;             // next tile (for prefetch)
    const int e1 = k1 >> 9;
    const int d1 = k1 & 511;
#pragma unroll
    for (int i = 0; i < 4; ++i) {
      int idx = tid + i * 256;
      int row = idx >> 3;
      int c   = (idx & 7) << 3;
      int gr  = blockM + row;
      const float* xp = &x[(size_t)gr * DD + d0 + c];
      const floatx4 x0 = *reinterpret_cast<const floatx4*>(xp);
      const floatx4 x1 = *reinterpret_cast<const floatx4*>(xp + 4);
      const float   gv = g[gr * EE + e];
      if (i == 0 && step + 1 < NSTEPS)
        __builtin_prefetch(&x[(size_t)gr * DD + d1 + c], 0, 1);  // global_prefetch
      bf16x4 h0 = __builtin_convertvector(x0 * gv, bf16x4);      // packed cvt RNE
      bf16x4 h1 = __builtin_convertvector(x1 * gv, bf16x4);
      *reinterpret_cast<bf16x8*>(&As[row * LDA + c]) =
          __builtin_shufflevector(h0, h1, 0, 1, 2, 3, 4, 5, 6, 7);
    }
#pragma unroll
    for (int i = 0; i < 2; ++i) {
      int idx = tid + i * 256;
      int n   = idx >> 3;
      int c   = (idx & 7) << 3;
      const float* ap = &alpha[((size_t)e * UU + (blockN + n)) * (size_t)DD + d0 + c];
      const floatx4 a0 = *reinterpret_cast<const floatx4*>(ap);
      const floatx4 a1 = *reinterpret_cast<const floatx4*>(ap + 4);
      if (i == 0 && step + 1 < NSTEPS)
        __builtin_prefetch(
            &alpha[((size_t)e1 * UU + (blockN + n)) * (size_t)DD + d1 + c], 0, 1);
      bf16x4 h0 = __builtin_convertvector(a0, bf16x4);
      bf16x4 h1 = __builtin_convertvector(a1, bf16x4);
      *reinterpret_cast<bf16x8*>(&Bs[n * LDB + c]) =
          __builtin_shufflevector(h0, h1, 0, 1, 2, 3, 4, 5, 6, 7);
    }
  }
}

// Two K=32 WMMA sub-steps over one staged BK=64 tile; each wave computes 32x32.
__device__ __forceinline__ void mma_tile(
    const u16* __restrict__ As, const u16* __restrict__ Bs,
    floatx8 acc[2][2], int wM, int wN, int lr, int half)
{
#pragma unroll
  for (int sub = 0; sub < 2; ++sub) {
    const int kk = sub * 32;
    bf16x16 a[2], b[2];
#pragma unroll
    for (int mt = 0; mt < 2; ++mt) {
      // A 16x32 bf16: lane<16 holds K 0-7 & 16-23; lane>=16 holds K 8-15 & 24-31
      const u16* p = &As[(wM * 32 + mt * 16 + lr) * LDA + kk + half * 8];
      bf16x8 a0 = *reinterpret_cast<const bf16x8*>(p);
      bf16x8 a1 = *reinterpret_cast<const bf16x8*>(p + 16);
      a[mt] = __builtin_shufflevector(a0, a1, 0, 1, 2, 3, 4, 5, 6, 7,
                                      8, 9, 10, 11, 12, 13, 14, 15);
    }
#pragma unroll
    for (int nt = 0; nt < 2; ++nt) {
      // B 32x16 bf16: lane<16 holds K 0-7 & 16-23; lane>=16 holds K 8-15 & 24-31
      const u16* p = &Bs[(wN * 32 + nt * 16 + lr) * LDB + kk + half * 8];
      bf16x8 b0 = *reinterpret_cast<const bf16x8*>(p);
      bf16x8 b1 = *reinterpret_cast<const bf16x8*>(p + 16);
      b[nt] = __builtin_shufflevector(b0, b1, 0, 1, 2, 3, 4, 5, 6, 7,
                                      8, 9, 10, 11, 12, 13, 14, 15);
    }
#pragma unroll
    for (int mt = 0; mt < 2; ++mt)
#pragma unroll
      for (int nt = 0; nt < 2; ++nt)
        acc[mt][nt] = __builtin_amdgcn_wmma_f32_16x16x32_bf16(
            false, a[mt], false, b[nt], (short)0, acc[mt][nt], false, false);
  }
}

__global__ __launch_bounds__(256)
void dense_expert_gemm(const float* __restrict__ x, const float* __restrict__ g,
                       const float* __restrict__ alpha, const float* __restrict__ beta,
                       float* __restrict__ out)
{
  __shared__ u16 As[2][BM * LDA];   // 2 * 18432 B
  __shared__ u16 Bs[2][BN * LDB];   // 2 *  9216 B   (total 55296 B LDS)

  const int tid  = threadIdx.x;
  const int lane = tid & 31;        // wave32
  const int wave = tid >> 5;        // 8 waves
  const int wM   = wave & 3;        // 4 waves along M -> 32 rows each
  const int wN   = wave >> 2;       // 2 waves along N -> 32 cols each
  const int lr   = lane & 15;
  const int half = lane >> 4;
  const int blockM = blockIdx.x * BM;
  const int blockN = blockIdx.y * BN;

  floatx8 acc[2][2] = {};

  stage_tiles(As[0], Bs[0], x, g, alpha, beta, blockM, blockN, 0, tid);
  __syncthreads();

  for (int s = 0; s < NSTEPS; ++s) {
    if (s + 1 < NSTEPS)
      stage_tiles(As[(s + 1) & 1], Bs[(s + 1) & 1], x, g, alpha, beta,
                  blockM, blockN, s + 1, tid);
    mma_tile(As[s & 1], Bs[s & 1], acc, wM, wN, lr, half);
    __syncthreads();
  }

  // C/D 16x16 f32 layout: VGPR i, lanes 0-15 -> M=i, lanes 16-31 -> M=i+8; N=lane&15
#pragma unroll
  for (int mt = 0; mt < 2; ++mt)
#pragma unroll
    for (int nt = 0; nt < 2; ++nt)
#pragma unroll
      for (int i = 0; i < 8; ++i) {
        const int row = blockM + wM * 32 + mt * 16 + half * 8 + i;
        const int col = blockN + wN * 32 + nt * 16 + lr;
        out[(size_t)row * UU + col] = acc[mt][nt][i];
      }
}

extern "C" void kernel_launch(void* const* d_in, const int* in_sizes, int n_in,
                              void* d_out, int out_size, void* d_ws, size_t ws_size,
                              hipStream_t stream) {
  const float* x     = (const float*)d_in[0];   // [8192, 512]
  const float* g     = (const float*)d_in[1];   // [8192, 8]
  const float* alpha = (const float*)d_in[2];   // [8, 512, 512]
  const float* beta  = (const float*)d_in[3];   // [8, 512]
  float* out = (float*)d_out;                   // [8192, 512]

  dim3 grid(BB / BM, UU / BN);                  // 64 x 8 = 512 workgroups
  dim3 block(256);                              // 8 wave32 waves
  dense_expert_gemm<<<grid, block, 0, stream>>>(x, g, alpha, beta, out);
}
